// FullAttention_28467043238016
// MI455X (gfx1250) — matrix operand order repaired and checked by reading, not compiled, not run on hardware
//
#include <hip/hip_runtime.h>

typedef _Float16 half2v  __attribute__((ext_vector_type(2)));
typedef __fp16   fp16x2  __attribute__((ext_vector_type(2)));
typedef _Float16 v8h     __attribute__((ext_vector_type(8)));
typedef _Float16 v16h    __attribute__((ext_vector_type(16)));
typedef float    v8f     __attribute__((ext_vector_type(8)));

namespace {
constexpr int kN = 2, kL = 4096, kS = 4096, kH = 8, kD = 64;
constexpr int SCHUNK = 64;            // S columns per pipeline stage
constexpr int NITER  = kS / SCHUNK;   // 64
constexpr int LSTR   = 72;            // halves per LDS row (64 + pad, 16B-aligned)
constexpr float kScale = 0.18033688f; // (1/sqrt(D)) * log2(e), folded into Q

union FragU {  // one 32-lane-striped WMMA operand (8 VGPRs per lane)
  v16h   v;
  v8h    h8[2];
  half2v h2[8];
};
union Pack8 {
  v8h    v;
  half2v h2[4];
};
}  // namespace

__device__ __forceinline__ half2v cvt_pk(float a, float b) {
  fp16x2 r = __builtin_amdgcn_cvt_pkrtz(a, b);
  union { fp16x2 x; half2v y; } u;
  u.x = r;
  return u.y;
}

__device__ __forceinline__ v8f wmma_f16(v16h a, v16h b, v8f c) {
  // D = A(16x32 f16) * B(32x16 f16) + C(16x16 f32)
  return __builtin_amdgcn_wmma_f32_16x16x32_f16(false, a, false, b, (short)0, c,
                                                false, false);
}

__global__ __launch_bounds__(256) void fa_fwd(const float* __restrict__ Q,
                                              const float* __restrict__ K,
                                              const float* __restrict__ V,
                                              float* __restrict__ O) {
  // double-buffered staging: K row-major f16, V transposed f16
  __shared__ __align__(16) _Float16 Ksh[2][SCHUNK * LSTR];
  __shared__ __align__(16) _Float16 VTsh[2][kD * LSTR];

  const int tid  = threadIdx.x;
  const int lane = tid & 31;
  const int wave = tid >> 5;
  const int m16  = lane & 15;
  const int hi   = lane >> 4;

  const int n  = blockIdx.z;
  const int h  = blockIdx.y;
  const int q0 = blockIdx.x * 128 + wave * 16;  // this wave's 16 query rows

  // ---- Q fragment: B operand of S^T = K x Q^T, pre-scaled by temp*log2(e).
  // B layout (32x16 f16): lane -> column q = lane&15, half h<8 -> d = 8*hi + h; h>=8 -> d = 8*hi + 16 + (h-8).
  FragU bq[2];
  {
    const float* qrow = Q + (((size_t)n * kL + q0 + m16) * kH + h) * kD;
#pragma unroll
    for (int f = 0; f < 2; ++f) {
      const float4* p4 = (const float4*)(qrow + f * 32 + hi * 8);
#pragma unroll
      for (int i = 0; i < 4; ++i) {
        float4 x = p4[i + (i & 2)];
        bq[f].h2[2 * i]     = cvt_pk(x.x * kScale, x.y * kScale);
        bq[f].h2[2 * i + 1] = cvt_pk(x.z * kScale, x.w * kScale);
      }
    }
  }

  // Output accumulators: O tile 16(q) x 64(d) = 4 C/D tiles.
  // C/D layout: lane -> col d = 16*dt + (lane&15), VGPR r -> row q = r + 8*hi.
  v8f acc[4];
  const v8f vzero = {};
#pragma unroll
  for (int dt = 0; dt < 4; ++dt) acc[dt] = vzero;

  float mrow = -3.0e38f;  // running row max (per query q = lane&15)
  float srow = 0.0f;      // running row sum

  // cooperative staging: thread -> (row 0..63, 16-wide d segment)
  const int ldrow = tid >> 2;
  const int ldcol = (tid & 3) * 16;
  const float* kptr =
      K + ((size_t)n * kS * kH + h) * kD + (size_t)ldrow * (kH * kD) + ldcol;
  const float* vptr =
      V + ((size_t)n * kS * kH + h) * kD + (size_t)ldrow * (kH * kD) + ldcol;

  // ---- prologue: pull chunk 0 into registers ----
  float4 kst[4], vst[4];
#pragma unroll
  for (int i = 0; i < 4; ++i) {
    kst[i] = ((const float4*)kptr)[i];
    vst[i] = ((const float4*)vptr)[i];
  }

  for (int it = 0; it < NITER; ++it) {
    const int buf = it & 1;

    // ---- stage registers -> LDS (f32 -> f16); V stored transposed ----
    {
      _Float16* krow = &Ksh[buf][ldrow * LSTR + ldcol];
      Pack8 a, b;
      a.h2[0] = cvt_pk(kst[0].x, kst[0].y);
      a.h2[1] = cvt_pk(kst[0].z, kst[0].w);
      a.h2[2] = cvt_pk(kst[1].x, kst[1].y);
      a.h2[3] = cvt_pk(kst[1].z, kst[1].w);
      b.h2[0] = cvt_pk(kst[2].x, kst[2].y);
      b.h2[1] = cvt_pk(kst[2].z, kst[2].w);
      b.h2[2] = cvt_pk(kst[3].x, kst[3].y);
      b.h2[3] = cvt_pk(kst[3].z, kst[3].w);
      *(v8h*)(krow)     = a.v;
      *(v8h*)(krow + 8) = b.v;

      const _Float16 vh[16] = {
          (_Float16)vst[0].x, (_Float16)vst[0].y, (_Float16)vst[0].z,
          (_Float16)vst[0].w, (_Float16)vst[1].x, (_Float16)vst[1].y,
          (_Float16)vst[1].z, (_Float16)vst[1].w, (_Float16)vst[2].x,
          (_Float16)vst[2].y, (_Float16)vst[2].z, (_Float16)vst[2].w,
          (_Float16)vst[3].x, (_Float16)vst[3].y, (_Float16)vst[3].z,
          (_Float16)vst[3].w};
#pragma unroll
      for (int i = 0; i < 16; ++i)
        VTsh[buf][(ldcol + i) * LSTR + ldrow] = vh[i];
    }

    // ---- issue next chunk's global loads (in flight across the compute) ----
    if (it + 1 < NITER) {
      const float* kn = kptr + (size_t)(it + 1) * SCHUNK * (kH * kD);
      const float* vn = vptr + (size_t)(it + 1) * SCHUNK * (kH * kD);
#pragma unroll
      for (int i = 0; i < 4; ++i) {
        kst[i] = ((const float4*)kn)[i];
        vst[i] = ((const float4*)vn)[i];
      }
    }

    __syncthreads();  // buf[it&1] visible; prior buffer's readers are done

    // ---- S^T = K_tile x Q^T : four 16x16 transposed score tiles ----
    // A layout (16x32 f16): lane -> row s = 16*tt + (lane&15);
    //   half h<8 -> d = 8*hi + h ; h>=8 -> d = 8*hi + 16 + (h-8).
    v8f st[4];
#pragma unroll
    for (int tt = 0; tt < 4; ++tt) {
      const _Float16* kr = &Ksh[buf][(tt * 16 + m16) * LSTR + hi * 8];
      FragU ka0, ka1;
      ka0.h8[0] = *(const v8h*)(kr);
      ka0.h8[1] = *(const v8h*)(kr + 16);
      ka1.h8[0] = *(const v8h*)(kr + 32);
      ka1.h8[1] = *(const v8h*)(kr + 48);
      v8f c = vzero;
      c = wmma_f16(ka0.v, bq[0].v, c);
      c = wmma_f16(ka1.v, bq[1].v, c);
      st[tt] = c;
    }

    // ---- online softmax over this 64-column slab (stats are in-lane) ----
    float p[32];
#pragma unroll
    for (int tt = 0; tt < 4; ++tt)
#pragma unroll
      for (int i = 0; i < 8; ++i) p[tt * 8 + i] = st[tt][i];
    float mx = p[0];
#pragma unroll
    for (int i = 1; i < 32; ++i) mx = fmaxf(mx, p[i]);
    mx = fmaxf(mx, __shfl_xor(mx, 16, 32));  // combine the two half-wave groups
    const float mnew = fmaxf(mrow, mx);
    // raw v_exp_f32: args <= 0; large-negative flush-to-zero is desired
    const float corr = __builtin_amdgcn_exp2f(mrow - mnew);
    mrow = mnew;
    float ps = 0.0f;
#pragma unroll
    for (int i = 0; i < 32; ++i) {
      p[i] = __builtin_amdgcn_exp2f(p[i] - mnew);
      ps += p[i];
    }
    ps += __shfl_xor(ps, 16, 32);
    srow = srow * corr + ps;

    // ---- P fragments: transposed-score C/D layout *is* the A layout ----
    FragU ap[2];
#pragma unroll
    for (int i = 0; i < 8; ++i) {
      ap[0].h2[i] = cvt_pk(p[2 * i], p[2 * i + 1]);          // s = 0..31
      ap[1].h2[i] = cvt_pk(p[16 + 2 * i], p[16 + 2 * i + 1]); // s = 32..63
    }

    // ---- rescale accumulators: broadcast corr to row-indexed layout ----
    const int ci = __float_as_int(corr);
    float cb[8];
#pragma unroll
    for (int r = 0; r < 8; ++r)
      cb[r] = __int_as_float(__builtin_amdgcn_ds_bpermute((hi * 8 + r) << 2, ci));
#pragma unroll
    for (int dt = 0; dt < 4; ++dt)
#pragma unroll
      for (int r = 0; r < 8; ++r) acc[dt][r] *= cb[r];

    // ---- O += P(16x64) x V(64x64) : 4 output tiles, K split in two ----
    // B layout: lane -> col d = 16*dt + (lane&15), half h<8 -> s = f*32 + 8*hi + h; h>=8 -> s = f*32 + 8*hi + 16 + (h-8).
#pragma unroll
    for (int dt = 0; dt < 4; ++dt) {
      const _Float16* vr = &VTsh[buf][(dt * 16 + m16) * LSTR + hi * 8];
      FragU bv0, bv1;
      bv0.h8[0] = *(const v8h*)(vr);
      bv0.h8[1] = *(const v8h*)(vr + 16);
      bv1.h8[0] = *(const v8h*)(vr + 32);
      bv1.h8[1] = *(const v8h*)(vr + 48);
      acc[dt] = wmma_f16(ap[0].v, bv0.v, acc[dt]);
      acc[dt] = wmma_f16(ap[1].v, bv1.v, acc[dt]);
    }
  }

  // ---- epilogue: divide by row sums, store ----
  const float inv = __builtin_amdgcn_rcpf(srow);
  const int ii = __float_as_int(inv);
  float ib[8];
#pragma unroll
  for (int r = 0; r < 8; ++r)
    ib[r] = __int_as_float(__builtin_amdgcn_ds_bpermute((hi * 8 + r) << 2, ii));
#pragma unroll
  for (int dt = 0; dt < 4; ++dt) {
#pragma unroll
    for (int r = 0; r < 8; ++r) {
      float* dst = O + (((size_t)n * kL + q0 + 8 * hi + r) * kH + h) * kD +
                   dt * 16 + m16;
      __builtin_nontemporal_store(acc[dt][r] * ib[r], dst);
    }
  }
}

extern "C" void kernel_launch(void* const* d_in, const int* in_sizes, int n_in,
                              void* d_out, int out_size, void* d_ws,
                              size_t ws_size, hipStream_t stream) {
  const float* Q = (const float*)d_in[0];
  const float* K = (const float*)d_in[1];
  const float* V = (const float*)d_in[2];
  float* O = (float*)d_out;
  dim3 grid(kL / 128, kH, kN);  // (32, 8, 2)
  fa_fwd<<<grid, dim3(256), 0, stream>>>(Q, K, V, O);
}
